// MHNGDNet_43104291783157
// MI455X (gfx1250) — matrix operand order repaired and checked by reading, not compiled, not run on hardware
//
#include <hip/hip_runtime.h>
#include <stdint.h>

typedef __attribute__((ext_vector_type(16))) _Float16 v16h;
typedef __attribute__((ext_vector_type(8)))  _Float16 v8h;
typedef __attribute__((ext_vector_type(8)))  float    v8f;
typedef __attribute__((ext_vector_type(2)))  float    v2f;
typedef __attribute__((ext_vector_type(8)))  uint32_t v8u;

#define MRR 128
#define NTT 128
#define SSAMP 32
#define ITERS 16
#define BSZ 512
#define BETA_C 0.9f
#define DQAM_C 0.31622776601683793720f

// ---------------- helpers ----------------
__device__ __forceinline__ uint32_t hash_u32(uint32_t x){
  x ^= x >> 16; x *= 0x7feb352du; x ^= x >> 15; x *= 0x846ca68bu; x ^= x >> 16; return x;
}
__device__ __forceinline__ float u01(uint32_t h){ return ((float)((h >> 8) + 1u)) * (1.0f/16777216.0f); }

__device__ __forceinline__ float qnear(float v){
  float t = v * (1.0f/DQAM_C);
  float l = (t < -2.0f) ? -3.0f : (t < 0.0f) ? -1.0f : (t < 2.0f) ? 1.0f : 3.0f;
  return l * DQAM_C;
}

__device__ __forceinline__ v16h cat16(v8h a, v8h b){
  return __builtin_shufflevector(a, b, 0,1,2,3,4,5,6,7,8,9,10,11,12,13,14,15);
}
__device__ __forceinline__ v16h neg16(v16h x){
  v8u u = __builtin_bit_cast(v8u, x);
  u = u ^ 0x80008000u;               // packed f16 sign flip
  return __builtin_bit_cast(v16h, u);
}

// in-place complex Gauss-Jordan inversion of 128x128 (Mr,Mi in LDS), 256 threads
__device__ __forceinline__ void gj_inv128(float* Mr, float* Mi, float* piv, int tid){
  for (int k = 0; k < 128; ++k){
    if (tid == 0){
      float a = Mr[k*128+k], c = Mi[k*128+k];
      float d = 1.0f / (a*a + c*c + 1e-30f);
      piv[0] = a*d; piv[1] = -c*d;
    }
    __syncthreads();
    float pr = piv[0], pi = piv[1];
    if (tid < 128 && tid != k){
      int j = tid; float a = Mr[k*128+j], c = Mi[k*128+j];
      Mr[k*128+j] = a*pr - c*pi; Mi[k*128+j] = a*pi + c*pr;
    }
    __syncthreads();
    {
      int i  = tid >> 1;
      int j0 = (tid & 1) * 64;
      if (i != k){
        float fr = Mr[i*128+k], fi = Mi[i*128+k];
        for (int j = j0; j < j0 + 64; ++j){
          if (j == k) continue;
          float ar = Mr[k*128+j], ai = Mi[k*128+j];
          Mr[i*128+j] -= fr*ar - fi*ai;
          Mi[i*128+j] -= fr*ai + fi*ar;
        }
      }
    }
    __syncthreads();
    if (tid < 128){
      if (tid != k){
        int i = tid; float fr = Mr[i*128+k], fi = Mi[i*128+k];
        Mr[i*128+k] = -(fr*pr - fi*pi);
        Mi[i*128+k] = -(fr*pi + fi*pr);
      } else { Mr[k*128+k] = pr; Mi[k*128+k] = pi; }
    }
    __syncthreads();
  }
}

// Complex GEMM, all state s-major:
//   Z[s][m] (s=0..31, m=0..127) = op(P)[128x128] * B  with B staged as Bs[s][k].
// OPL=false: P is row-major f16 in LDS (A), element (m,k) at P[m*128+k].
// OPL=true : P is a pre-swizzled WMMA-operand plane in GLOBAL memory:
//            lane-contiguous 16-half packets indexed by ((mt*4+kc)*32+lane)*16.
// 8 waves: wave w owns M-tile w; N-tiles 0..1 (samplers); K in 4 chunks of 32.
template<bool OPL>
__device__ __forceinline__ void cgemm_s(const _Float16* Pr, const _Float16* Pi,
                                        const _Float16* Bs_r, const _Float16* Bs_i,
                                        float* Zs_r, float* Zs_i, int tid)
{
  const int wave = tid >> 5, lane = tid & 31;
  const int ml = lane & 15, hi = lane >> 4;
  const int mt = wave;
  for (int nt = 0; nt < 2; ++nt){
    v8f cre = {}; v8f cim = {};
    const int n = nt*16 + ml;
    for (int kc = 0; kc < 4; ++kc){
      v16h ar, ai, br, bi;
      if (OPL){
        const int base = ((mt*4 + kc)*32 + lane)*16;
        ar = cat16(*(const v8h*)(Pr + base), *(const v8h*)(Pr + base + 8));
        ai = cat16(*(const v8h*)(Pi + base), *(const v8h*)(Pi + base + 8));
      } else {
        const int base = (mt*16 + ml)*128 + kc*32 + hi*8;   // k = kc*32+hi*8+0..7 / +16
        ar = cat16(*(const v8h*)(Pr + base), *(const v8h*)(Pr + base + 16));
        ai = cat16(*(const v8h*)(Pi + base), *(const v8h*)(Pi + base + 16));
      }
      {
        const int bb = n*128 + kc*32 + hi*8;                // k = kc*32+hi*8+0..7 / +16
        br = cat16(*(const v8h*)(Bs_r + bb), *(const v8h*)(Bs_r + bb + 16));
        bi = cat16(*(const v8h*)(Bs_i + bb), *(const v8h*)(Bs_i + bb + 16));
      }
      v16h nai = neg16(ai);
      cre = __builtin_amdgcn_wmma_f32_16x16x32_f16(false, ar,  false, br, (short)0, cre, false, false);
      cre = __builtin_amdgcn_wmma_f32_16x16x32_f16(false, nai, false, bi, (short)0, cre, false, false);
      cim = __builtin_amdgcn_wmma_f32_16x16x32_f16(false, ar,  false, bi, (short)0, cim, false, false);
      cim = __builtin_amdgcn_wmma_f32_16x16x32_f16(false, ai,  false, br, (short)0, cim, false, false);
    }
    // C/D rows contiguous in M -> one v8f (2x ds_store_b128) per plane
    const int m0 = mt*16 + hi*8;
    *(v8f*)(Zs_r + n*128 + m0) = cre;
    *(v8f*)(Zs_i + n*128 + m0) = cim;
  }
}

// ---------------- kernels ----------------
__global__ void zero_kernel(float* p, int n){
  int t = threadIdx.x;
  if (t < n) p[t] = 0.0f;
}

// Per-batch prep:
//  - f16 row-major A planes (for async LDS fill in sampler)
//  - A^H and covar=inv(A)*colnorm written as pre-swizzled f16 WMMA-operand planes
//  - G = inv(A^H A + nv I) (f32 WMMA for A^H A), MMSE init, quantize, rn0/step0
__global__ __launch_bounds__(256) void prep_kernel(
    const float* __restrict__ Are_g, const float* __restrict__ Aim_g,
    const float* __restrict__ yre_g, const float* __restrict__ yim_g,
    const float* __restrict__ nv_g,
    _Float16* __restrict__ a16r_g, _Float16* __restrict__ a16i_g,
    _Float16* __restrict__ ahopr_g, _Float16* __restrict__ ahopi_g,
    _Float16* __restrict__ cvopr_g, _Float16* __restrict__ cvopi_g,
    float* __restrict__ xh0r_g, float* __restrict__ xh0i_g,
    float* __restrict__ rn0_g, float* __restrict__ st0_g)
{
  extern __shared__ float sm[];
  float* Ar = sm;            // 16384
  float* Ai = Ar + 16384;
  float* Mr = Ai + 16384;
  float* Mi = Mr + 16384;
  float* aux = Mi + 16384;
  float* yre = aux;          // 128
  float* yim = yre + 128;
  float* tre = yim + 128;
  float* tim = tre + 128;
  float* red = tim + 128;    // 256
  float* piv = red + 256;    // 2

  const int b = blockIdx.x, tid = threadIdx.x;
  const float nv = nv_g[0];
  const float* Abr = Are_g + (size_t)b * 16384;
  const float* Abi = Aim_g + (size_t)b * 16384;

  for (int e = tid; e < 16384; e += 256){
    float r = Abr[e], i = Abi[e];
    Ar[e] = r; Ai[e] = i; Mr[e] = r; Mi[e] = i;
  }
  if (tid < 128){ yre[tid] = yre_g[b*128 + tid]; yim[tid] = yim_g[b*128 + tid]; }
  __syncthreads();

  // row-major f16 A planes + A^H operand-layout planes
  for (int e = tid; e < 16384; e += 256){
    a16r_g[(size_t)b*16384 + e] = (_Float16)Ar[e];
    a16i_g[(size_t)b*16384 + e] = (_Float16)Ai[e];
    // operand-layout decode: e = ((mt*4+kc)*32+lane)*16 + j
    int j    = e & 15;
    int lane = (e >> 4) & 31;
    int kc   = (e >> 9) & 3;
    int mt   = e >> 11;
    int ml = lane & 15, hi = lane >> 4;
    int k = kc*32 + ((j < 8) ? (hi*8 + j) : (16 + hi*8 + (j - 8)));
    int m = mt*16 + ml;
    ahopr_g[(size_t)b*16384 + e] = (_Float16)( Ar[k*128 + m]);   // A^H[m][k] = conj(A[k][m])
    ahopi_g[(size_t)b*16384 + e] = (_Float16)(-Ai[k*128 + m]);
  }
  __syncthreads();

  // ---- Ainv = inv(A) (in Mr/Mi) ----
  gj_inv128(Mr, Mi, piv, tid);
  if (tid < 128){
    int j = tid; float s = 0.0f;
    for (int i = 0; i < 128; ++i){ float a = Mr[i*128+j], c = Mi[i*128+j]; s += a*a + c*c; }
    red[j] = rsqrtf(s + 1e-30f);
  }
  __syncthreads();
  // covar operand-layout planes: covar[m][k] = Ainv[m][k] * colnorm[k]
  for (int e = tid; e < 16384; e += 256){
    int j    = e & 15;
    int lane = (e >> 4) & 31;
    int kc   = (e >> 9) & 3;
    int mt   = e >> 11;
    int ml = lane & 15, hi = lane >> 4;
    int k = kc*32 + ((j < 8) ? (hi*8 + j) : (16 + hi*8 + (j - 8)));
    int m = mt*16 + ml;
    float cn = red[k];
    cvopr_g[(size_t)b*16384 + e] = (_Float16)(Mr[m*128 + k] * cn);
    cvopi_g[(size_t)b*16384 + e] = (_Float16)(Mi[m*128 + k] * cn);
  }
  __syncthreads();

  // ---- M = A^H A + nv I  via f32 WMMA (feeds inversion -> keep f32) ----
  {
    const int wave = tid >> 5, lane = tid & 31;
    const int ml = lane & 15, hi = lane >> 4;
    const int mt = wave;
    for (int nt = 0; nt < 8; ++nt){
      v8f cre = {}; v8f cim = {};
      const int m = mt*16 + ml, n = nt*16 + ml;
      for (int kc = 0; kc < 32; ++kc){
        int kb = kc*4 + hi*2;
        v2f pr, pi, npi, br, bi;
        pr[0] = Ar[kb*128 + m]; pr[1] = Ar[(kb+1)*128 + m];
        pi[0] = Ai[kb*128 + m]; pi[1] = Ai[(kb+1)*128 + m];
        npi[0] = -pi[0]; npi[1] = -pi[1];
        br[0] = Ar[kb*128 + n]; br[1] = Ar[(kb+1)*128 + n];
        bi[0] = Ai[kb*128 + n]; bi[1] = Ai[(kb+1)*128 + n];
        cre = __builtin_amdgcn_wmma_f32_16x16x4_f32(false, pr,  false, br, (short)0, cre, false, false);
        cre = __builtin_amdgcn_wmma_f32_16x16x4_f32(false, pi,  false, bi, (short)0, cre, false, false);
        cim = __builtin_amdgcn_wmma_f32_16x16x4_f32(false, pr,  false, bi, (short)0, cim, false, false);
        cim = __builtin_amdgcn_wmma_f32_16x16x4_f32(false, npi, false, br, (short)0, cim, false, false);
      }
#pragma unroll
      for (int r = 0; r < 8; ++r){
        int mm = mt*16 + hi*8 + r, nn = nt*16 + ml;
        Mr[mm*128 + nn] = cre[r] + ((mm == nn) ? nv : 0.0f);
        Mi[mm*128 + nn] = cim[r];
      }
    }
  }
  __syncthreads();

  gj_inv128(Mr, Mi, piv, tid);   // G = inv(A^H A + nv I)

  // ---- t = A^H y ----
  if (tid < 128){
    int m = tid; float sr = 0, si = 0;
    for (int k = 0; k < 128; ++k){
      float ar = Ar[k*128+m], ai = -Ai[k*128+m];
      float yr = yre[k], yi = yim[k];
      sr += ar*yr - ai*yi; si += ar*yi + ai*yr;
    }
    tre[m] = sr; tim[m] = si;
  }
  __syncthreads();
  // ---- x_mmse = G t ; quantize -> xhat0 ----
  if (tid < 128){
    int m = tid; float sr = 0, si = 0;
    for (int j = 0; j < 128; ++j){
      float gr = Mr[m*128+j], gi = Mi[m*128+j];
      sr += gr*tre[j] - gi*tim[j];
      si += gr*tim[j] + gi*tre[j];
    }
    float qr = qnear(sr), qi = qnear(si);
    xh0r_g[b*128 + m] = qr; xh0i_g[b*128 + m] = qi;
  }
  __syncthreads();
  if (tid < 128){ tre[tid] = xh0r_g[b*128 + tid]; tim[tid] = xh0i_g[b*128 + tid]; }
  __syncthreads();
  // ---- r0 = y - A xhat0 ; rn0, step0 ----
  float part = 0.0f;
  if (tid < 128){
    int i = tid; float sr = 0, si = 0;
    for (int j = 0; j < 128; ++j){
      float ar = Ar[i*128+j], ai = Ai[i*128+j];
      sr += ar*tre[j] - ai*tim[j];
      si += ar*tim[j] + ai*tre[j];
    }
    float rr = yre[i] - sr, ri = yim[i] - si;
    part = rr*rr + ri*ri;
  }
  red[tid] = part; __syncthreads();
  for (int w = 128; w > 0; w >>= 1){ if (tid < w) red[tid] += red[tid + w]; __syncthreads(); }
  if (tid == 0){
    float rn0 = red[0];
    rn0_g[b] = rn0;
    float st = sqrtf(rn0 / 128.0f);
    st0_g[b] = (st > DQAM_C) ? st : DQAM_C;
  }
}

// LDS layout for sampler (bytes):
//  halves: Ar16,Ai16 (2*16384) + Bs_r,Bs_i (2*4096) = 40960 h = 81920 B
//  floats: Zr,Zi,XHr,XHi,MMr,MMi,XPr,XPi,XSr,XSi = 40960 f = 163840 B ; aux 1024 f
#define SMP_HALF_BYTES 81920
#define SMP_BYTES      (SMP_HALF_BYTES + (40960 + 1024) * 4)

__global__ __launch_bounds__(256) void samp_kernel(
    const float* __restrict__ xre_g, const float* __restrict__ xim_g,
    const float* __restrict__ yre_g, const float* __restrict__ yim_g,
    const float* __restrict__ nv_g,  const float* __restrict__ lr_g,
    const _Float16* __restrict__ a16r_g, const _Float16* __restrict__ a16i_g,
    const _Float16* __restrict__ ahopr_g, const _Float16* __restrict__ ahopi_g,
    const _Float16* __restrict__ cvopr_g, const _Float16* __restrict__ cvopi_g,
    const float* __restrict__ xh0r_g, const float* __restrict__ xh0i_g,
    const float* __restrict__ rn0_g,  const float* __restrict__ st0_g,
    float* __restrict__ out_xre, float* __restrict__ out_xim,
    float* __restrict__ out_loss, float* __restrict__ out_mse)
{
  extern __shared__ char smraw[];
  _Float16* Ar16 = (_Float16*)smraw;          // [128][128] row-major
  _Float16* Ai16 = Ar16 + 16384;
  _Float16* Bs_r = Ai16 + 16384;              // [32][128] s-major staging
  _Float16* Bs_i = Bs_r + 4096;
  float* Zr  = (float*)(smraw + SMP_HALF_BYTES);  // [32][128] GEMM out, s-major
  float* Zi  = Zr  + 4096;
  float* XHr = Zi  + 4096;  float* XHi = XHr + 4096;   // xhat
  float* MMr = XHi + 4096;  float* MMi = MMr + 4096;   // momentum
  float* XPr = MMi + 4096;  float* XPi = XPr + 4096;   // x_prop
  float* XSr = XPi + 4096;  float* XSi = XSr + 4096;   // x_sv
  float* aux = XSi + 4096;
  float* yre = aux;           float* yim = yre + 128;
  float* x0r = yim + 128;     float* x0i = x0r + 128;
  float* rn   = x0i + 128;    // 32 each
  float* rnsv = rn   + 32;
  float* stp  = rnsv + 32;
  float* accm = stp  + 32;
  float* svm  = accm + 32;
  float* red  = svm  + 32;    // 256
  float* bsel = red + 256;    // 1

  const int b = blockIdx.x, tid = threadIdx.x;

  // ---- async DMA of the f16 A planes straight into LDS (CDNA5 async-to-LDS path) ----
  {
    uint64_t gr = (uint64_t)(uintptr_t)(a16r_g + (size_t)b * 16384);
    uint64_t gi = (uint64_t)(uintptr_t)(a16i_g + (size_t)b * 16384);
    uint32_t lr_ = (uint32_t)(uintptr_t)(void*)Ar16;   // low 32 bits of flat-LDS = LDS offset
    uint32_t li_ = (uint32_t)(uintptr_t)(void*)Ai16;
    for (int e = tid * 16; e < 16384; e += 256 * 16){  // 16 halves = 32B per lane per plane? no: 2x16B
      uint32_t off = (uint32_t)e * 2u;                 // byte offset
      asm volatile("global_load_async_to_lds_b128 %0, %1, %2"
                   :: "v"(lr_ + off), "v"(off), "s"(gr) : "memory");
      asm volatile("global_load_async_to_lds_b128 %0, %1, %2"
                   :: "v"(lr_ + off + 16), "v"(off + 16), "s"(gr) : "memory");
      asm volatile("global_load_async_to_lds_b128 %0, %1, %2"
                   :: "v"(li_ + off), "v"(off), "s"(gi) : "memory");
      asm volatile("global_load_async_to_lds_b128 %0, %1, %2"
                   :: "v"(li_ + off + 16), "v"(off + 16), "s"(gi) : "memory");
    }
    asm volatile("s_wait_asynccnt 0" ::: "memory");
  }

  if (tid < 128){
    yre[tid] = yre_g[b*128 + tid]; yim[tid] = yim_g[b*128 + tid];
    x0r[tid] = xh0r_g[b*128 + tid]; x0i[tid] = xh0i_g[b*128 + tid];
  }
  if (tid < 32){
    float r0 = rn0_g[b], s0 = st0_g[b];
    rn[tid] = r0; rnsv[tid] = r0; stp[tid] = s0;
  }
  __syncthreads();
  for (int e = tid; e < 4096; e += 256){
    int i = e & 127;
    XHr[e] = x0r[i]; XHi[e] = x0i[i];
    XSr[e] = x0r[i]; XSi[e] = x0i[i];
    MMr[e] = 0.0f;   MMi[e] = 0.0f;
  }
  __syncthreads();
  if (tid < 128){ x0r[tid] = xre_g[b*128 + tid]; x0i[tid] = xim_g[b*128 + tid]; } // true x
  __syncthreads();

  const _Float16* AHr = ahopr_g + (size_t)b * 16384;
  const _Float16* AHi = ahopi_g + (size_t)b * 16384;
  const _Float16* CVr = cvopr_g + (size_t)b * 16384;
  const _Float16* CVi = cvopi_g + (size_t)b * 16384;

  const float nv = nv_g[0];
  const float inv2nv = 0.5f / nv;

  for (int t = 0; t < ITERS; ++t){
    const float lrt = lr_g[t];

    // stage y_grad = xhat + beta*mom (f16, s-major: same index space)
    for (int e = tid; e < 4096; e += 256){
      Bs_r[e] = (_Float16)(XHr[e] + BETA_C*MMr[e]);
      Bs_i[e] = (_Float16)(XHi[e] + BETA_C*MMi[e]);
    }
    __syncthreads();
    cgemm_s<false>(Ar16, Ai16, Bs_r, Bs_i, Zr, Zi, tid);      // Z = A @ y_grad
    __syncthreads();

    // stage W = y - Z
    for (int e = tid; e < 4096; e += 256){
      int i = e & 127;
      Bs_r[e] = (_Float16)(yre[i] - Zr[e]);
      Bs_i[e] = (_Float16)(yim[i] - Zi[e]);
    }
    __syncthreads();
    cgemm_s<true>(AHr, AHi, Bs_r, Bs_i, Zr, Zi, tid);         // Z = A^H @ W  (operand plane from L2)
    __syncthreads();

    // mom = beta*mom + lr*Z  (z_grad = xhat + mom)
    for (int e = tid; e < 4096; e += 256){
      MMr[e] = BETA_C*MMr[e] + lrt*Zr[e];
      MMi[e] = BETA_C*MMi[e] + lrt*Zi[e];
    }
    // stage v ~ CN(0,1)
    for (int e = tid; e < 4096; e += 256){
      int i = e & 127, s = e >> 7;
      uint32_t base = ((uint32_t)t * 73856093u) ^ ((uint32_t)s * 19349663u)
                    ^ ((uint32_t)b * 83492791u) ^ ((uint32_t)i * 2654435761u);
      float u1 = u01(hash_u32(base + 1u));
      float u2 = u01(hash_u32(base + 2u));
      float rad = sqrtf(-2.0f * __logf(u1)) * 0.70710678f;
      float ang = 6.2831853f * u2;
      Bs_r[e] = (_Float16)(rad * __cosf(ang));
      Bs_i[e] = (_Float16)(rad * __sinf(ang));
    }
    __syncthreads();
    cgemm_s<true>(CVr, CVi, Bs_r, Bs_i, Zr, Zi, tid);         // Z = covar @ v (operand plane from L2)
    __syncthreads();

    // x_prop = quant(z_grad + step*Z)
    for (int e = tid; e < 4096; e += 256){
      int s = e >> 7; float st = stp[s];
      float qr = qnear(XHr[e] + MMr[e] + st*Zr[e]);
      float qi = qnear(XHi[e] + MMi[e] + st*Zi[e]);
      XPr[e] = qr; XPi[e] = qi;
      Bs_r[e] = (_Float16)qr; Bs_i[e] = (_Float16)qi;
    }
    __syncthreads();
    cgemm_s<false>(Ar16, Ai16, Bs_r, Bs_i, Zr, Zi, tid);      // Z = A @ x_prop
    __syncthreads();

    // rn_prop per sampler
    {
      int s = tid >> 3, ch = tid & 7; float p = 0.0f;
      for (int i = ch*16; i < ch*16 + 16; ++i){
        float rr = yre[i] - Zr[s*128 + i], ri = yim[i] - Zi[s*128 + i];
        p += rr*rr + ri*ri;
      }
      red[tid] = p;
    }
    __syncthreads();
    if (tid < 32){
      float rnp = 0.0f;
      for (int c = 0; c < 8; ++c) rnp += red[tid*8 + c];
      float sv = (rnp < rnsv[tid]) ? 1.0f : 0.0f;
      uint32_t hu = hash_u32(((uint32_t)t * 2246822519u) ^ ((uint32_t)tid * 3266489917u)
                             ^ ((uint32_t)b * 668265263u));
      float pu = u01(hu);
      float lacc = -(rnp - rn[tid]) * inv2nv;
      if (lacc > 0.0f) lacc = 0.0f;
      float acc = (__expf(lacc) >= pu) ? 1.0f : 0.0f;
      svm[tid] = sv; accm[tid] = acc;
      if (sv > 0.5f) rnsv[tid] = rnp;
      if (acc > 0.5f){
        rn[tid] = rnp;
        float stn = sqrtf(rnp / 128.0f);
        stp[tid] = (stn > DQAM_C) ? stn : DQAM_C;
      }
    }
    __syncthreads();

    // conditional state updates + mse partial
    float msep = 0.0f;
    for (int e = tid; e < 4096; e += 256){
      int i = e & 127, s = e >> 7;
      if (svm[s]  > 0.5f){ XSr[e] = XPr[e]; XSi[e] = XPi[e]; }
      if (accm[s] > 0.5f){ XHr[e] = XPr[e]; XHi[e] = XPi[e]; }
      float dr = XHr[e] - x0r[i], di = XHi[e] - x0i[i];
      msep += dr*dr + di*di;
    }
    __syncthreads();
    red[tid] = msep; __syncthreads();
    for (int w = 128; w > 0; w >>= 1){ if (tid < w) red[tid] += red[tid + w]; __syncthreads(); }
    if (tid == 0) atomicAdd(&out_mse[t], red[0] * (1.0f / (float)(BSZ * SSAMP)));
    __syncthreads();
  }

  // best sampler + loss
  if (tid == 0){
    float best = rnsv[0]; int bi = 0;
    float lsum = sqrtf(rnsv[0]);
    for (int s = 1; s < 32; ++s){
      lsum += sqrtf(rnsv[s]);
      if (rnsv[s] < best){ best = rnsv[s]; bi = s; }
    }
    bsel[0] = (float)bi;
    atomicAdd(out_loss, lsum * (1.0f / (float)BSZ));
  }
  __syncthreads();
  int sb = (int)bsel[0];
  if (tid < 128){
    out_xre[b*128 + tid] = XSr[sb*128 + tid];
    out_xim[b*128 + tid] = XSi[sb*128 + tid];
  }
}

// ---------------- host launcher ----------------
extern "C" void kernel_launch(void* const* d_in, const int* in_sizes, int n_in,
                              void* d_out, int out_size, void* d_ws, size_t ws_size,
                              hipStream_t stream)
{
  (void)in_sizes; (void)n_in; (void)out_size; (void)ws_size;
  const float* xre = (const float*)d_in[0];
  const float* xim = (const float*)d_in[1];
  const float* yre = (const float*)d_in[2];
  const float* yim = (const float*)d_in[3];
  const float* Are = (const float*)d_in[4];
  const float* Aim = (const float*)d_in[5];
  const float* nv  = (const float*)d_in[6];
  const float* lr  = (const float*)d_in[7];

  char* ws = (char*)d_ws;
  const size_t planeB = (size_t)BSZ * 16384 * sizeof(_Float16);
  _Float16* a16r = (_Float16*)ws;  ws += planeB;
  _Float16* a16i = (_Float16*)ws;  ws += planeB;
  _Float16* ahopr = (_Float16*)ws; ws += planeB;
  _Float16* ahopi = (_Float16*)ws; ws += planeB;
  _Float16* cvopr = (_Float16*)ws; ws += planeB;
  _Float16* cvopi = (_Float16*)ws; ws += planeB;
  float* xh0r = (float*)ws; ws += (size_t)BSZ * 128 * sizeof(float);
  float* xh0i = (float*)ws; ws += (size_t)BSZ * 128 * sizeof(float);
  float* rn0  = (float*)ws; ws += (size_t)BSZ * sizeof(float);
  float* st0  = (float*)ws; ws += (size_t)BSZ * sizeof(float);

  float* out      = (float*)d_out;
  float* out_xre  = out;
  float* out_xim  = out + (size_t)BSZ * NTT;
  float* out_loss = out + (size_t)2 * BSZ * NTT;
  float* out_mse  = out_loss + 1;

  zero_kernel<<<1, 32, 0, stream>>>(out_loss, 1 + ITERS);

  const size_t prepLds = (size_t)4 * 16384 * sizeof(float) + 4096;
  prep_kernel<<<BSZ, 256, prepLds, stream>>>(Are, Aim, yre, yim, nv,
                                             a16r, a16i, ahopr, ahopi, cvopr, cvopi,
                                             xh0r, xh0i, rn0, st0);

  samp_kernel<<<BSZ, 256, (size_t)SMP_BYTES, stream>>>(
      xre, xim, yre, yim, nv, lr,
      a16r, a16i, ahopr, ahopi, cvopr, cvopi,
      xh0r, xh0i, rn0, st0,
      out_xre, out_xim, out_loss, out_mse);
}
